// OnlineTripletLoss_88948772700362
// MI455X (gfx1250) — matrix operand order repaired and checked by reading, not compiled, not run on hardware
//
#include <hip/hip_runtime.h>
#include <hip/hip_bf16.h>

typedef __attribute__((ext_vector_type(16))) _Float16 v16h;
typedef __attribute__((ext_vector_type(8)))  _Float16 v8h;
typedef __attribute__((ext_vector_type(4)))  _Float16 v4h;
typedef __attribute__((ext_vector_type(8)))  float    v8f;
typedef __attribute__((ext_vector_type(4)))  float    v4f;
typedef __attribute__((ext_vector_type(4)))  int      v4i;

#define BATCH     4096
#define DIM       512
#define M_TILE    64
#define COL_SLICES 8
#define COLS_PER_SLICE (BATCH / COL_SLICES)   // 512

// ---- CDNA5 async global->LDS copy (guarded; falls back to sync copy) ----
#if defined(__has_builtin)
#  if __has_builtin(__builtin_amdgcn_global_load_async_to_lds_b128) && \
      __has_builtin(__builtin_amdgcn_s_wait_asynccnt)
#    define USE_ASYNC_LDS 1
#  endif
#endif
#ifndef USE_ASYNC_LDS
#  define USE_ASYNC_LDS 0
#endif

#define AS1 __attribute__((address_space(1)))
#define AS3 __attribute__((address_space(3)))

__device__ __forceinline__ AS1 v4i* to_global_v4i(const void* p) {
    return (AS1 v4i*)(unsigned long long)(uintptr_t)p;
}
__device__ __forceinline__ AS3 v4i* to_lds_v4i(const void* p) {
    // generic LDS pointer: low 32 bits are the LDS byte offset (aperture rule)
    return (AS3 v4i*)(unsigned)(uintptr_t)p;
}

// -------------------------------------------------------------------------
// Kernel 1: f32 -> f16 conversion, f32 squared row norms, sentinel init.
// -------------------------------------------------------------------------
__global__ __launch_bounds__(128)
void prep_kernel(const float* __restrict__ feat, _Float16* __restrict__ fh,
                 float* __restrict__ sq, int* __restrict__ d_ap, int* __restrict__ d_an) {
    const int row = blockIdx.x;
    const int tid = threadIdx.x;
    const float* src = feat + (size_t)row * DIM;
    _Float16*   dst = fh   + (size_t)row * DIM;

    v4f x = *(const v4f*)(src + tid * 4);
    v4h h;
    h.x = (_Float16)x.x; h.y = (_Float16)x.y; h.z = (_Float16)x.z; h.w = (_Float16)x.w;
    *(v4h*)(dst + tid * 4) = h;

    float s = x.x * x.x + x.y * x.y + x.z * x.z + x.w * x.w;

    __shared__ float red[128];
    red[tid] = s;
    __syncthreads();
    #pragma unroll
    for (int off = 64; off > 0; off >>= 1) {
        if (tid < off) red[tid] += red[tid + off];
        __syncthreads();
    }
    if (tid == 0) {
        sq[row]   = red[0];
        d_ap[row] = __float_as_int(-1.0f);   // sentinel: no positive (negative int)
        d_an[row] = __float_as_int(1e30f);   // sentinel: no negative
    }
}

// -------------------------------------------------------------------------
// Kernel 2: fused WMMA Gram-matrix tile + batch-hard mining epilogue.
// grid = (BATCH/M_TILE, COL_SLICES), block = 256 (8 waves).
// __launch_bounds__(256,4): cap at 256 VGPRs -> no VGPR-MSB mode switching,
// 4 waves/SIMD to hide L2 latency behind the WMMA pipe.
// -------------------------------------------------------------------------
__global__ __launch_bounds__(256, 4)
void miner_kernel(const _Float16* __restrict__ fh, const float* __restrict__ sq,
                  const int* __restrict__ label,
                  int* __restrict__ d_ap, int* __restrict__ d_an) {
    // A tile: 64 rows x 512 halves, XOR-swizzled 16B chunks (conflict-free).
    __shared__ __align__(16) _Float16 As[M_TILE * DIM];   // 64 KB

    const int rowBase = blockIdx.x * M_TILE;
    const int colBase = blockIdx.y * COLS_PER_SLICE;
    const int tid     = threadIdx.x;

    // Cooperative staging: 4096 x 16B chunks, 16 per thread.
    #pragma unroll
    for (int j = 0; j < 16; ++j) {
        const int c  = tid + 256 * j;
        const int r  = c >> 6;
        const int cc = c & 63;
        const _Float16* gsrc = fh + (size_t)(rowBase + r) * DIM + cc * 8;
        _Float16*       ldst = &As[r * DIM + ((cc ^ (r & 15)) << 3)];
#if USE_ASYNC_LDS
        __builtin_amdgcn_global_load_async_to_lds_b128(to_global_v4i(gsrc), to_lds_v4i(ldst), 0, 0);
#else
        *(uint4*)ldst = *(const uint4*)gsrc;
#endif
    }
#if USE_ASYNC_LDS
    __builtin_amdgcn_s_wait_asynccnt(0);
#endif
    __syncthreads();

    const int wave    = tid >> 5;
    const int lane    = tid & 31;
    const int laneLo  = lane & 15;
    const int laneHi  = lane >> 4;
    const int rowSub  = wave & 3;
    const int colHalf = wave >> 2;

    const int rowStart = rowBase + rowSub * 16;
    const int rowL     = rowSub * 16;
    const int aBase    = (rowL + laneLo) * DIM;   // LDS row base for A fragments

    float dmax[8], dmin[8];
    #pragma unroll
    for (int t = 0; t < 8; ++t) { dmax[t] = -1.0f; dmin[t] = 1e30f; }

    for (int chunk = 0; chunk < 2; ++chunk) {
        const int nBase0 = colBase + colHalf * 256 + chunk * 128;  // 8 N-tiles of 16

        // Hoisted B base pointers: K advance becomes an immediate offset.
        const _Float16* bptr[8];
        #pragma unroll
        for (int n = 0; n < 8; ++n)
            bptr[n] = fh + (size_t)(nBase0 + n * 16 + laneLo) * DIM + laneHi * 16;

        v8f acc[8] = {};

        #pragma unroll 4
        for (int kc = 0; kc < 16; ++kc) {              // K = 512, steps of 32
            // A fragment (16-bit A 16x32 layout): two 16B LDS chunks.
            union { v16h v; v8h h[2]; } a;
            const int c0 = 4 * kc + 2 * laneHi;
            a.h[0] = *(const v8h*)(&As[aBase + (((c0    ) ^ laneLo) << 3)]);
            a.h[1] = *(const v8h*)(&As[aBase + (((c0 + 1) ^ laneLo) << 3)]);

            #pragma unroll
            for (int n = 0; n < 8; ++n) {
                // B fragment (32x16): column = lane%16, K-half = lane/16;
                // base + kc*64B immediate offset -> no hot-loop address math.
                const v16h b = *(const v16h*)(bptr[n] + kc * 32);
                acc[n] = __builtin_amdgcn_wmma_f32_16x16x32_f16(
                    false, a.v, false, b, (short)0, acc[n], false, false);
            }
        }

        // Epilogue: distances + masked running max/min (metadata loaded here,
        // not across the GEMM loop, to keep it off the hot-loop live range).
        int   lm[8];
        float sqm[8];
        #pragma unroll
        for (int t = 0; t < 8; ++t) {
            const int m = rowStart + t + 8 * laneHi;
            lm[t]  = label[m];
            sqm[t] = sq[m];
        }
        #pragma unroll
        for (int n = 0; n < 8; ++n) {
            const int   col = nBase0 + n * 16 + laneLo;
            const int   ln  = label[col];
            const float sqn = sq[col];
            #pragma unroll
            for (int t = 0; t < 8; ++t) {
                const int m = rowStart + t + 8 * laneHi;
                float dist = sqm[t] + sqn - 2.0f * acc[n][t];
                dist = fmaxf(dist, 0.0f);
                const bool same = (lm[t] == ln);
                if (same && (m != col)) dmax[t] = fmaxf(dmax[t], dist);
                if (!same)              dmin[t] = fminf(dmin[t], dist);
            }
        }
    }

    // Butterfly reduce across the 16 lanes sharing each row, then publish via
    // order-independent integer atomics (dist >= 0 -> float bits monotone).
    #pragma unroll
    for (int t = 0; t < 8; ++t) {
        float vmax = dmax[t], vmin = dmin[t];
        #pragma unroll
        for (int off = 1; off < 16; off <<= 1) {
            vmax = fmaxf(vmax, __shfl_xor(vmax, off, 16));
            vmin = fminf(vmin, __shfl_xor(vmin, off, 16));
        }
        if (laneLo == 0) {
            const int m = rowStart + t + 8 * laneHi;
            if (vmax >= 0.0f)  atomicMax(&d_ap[m], __float_as_int(vmax));
            if (vmin < 1e30f)  atomicMin(&d_an[m], __float_as_int(vmin));
        }
    }
}

// -------------------------------------------------------------------------
// Kernel 3: per-anchor softplus + validity, reduce to scalar loss.
// -------------------------------------------------------------------------
__global__ __launch_bounds__(1024)
void finalize_kernel(const int* __restrict__ d_ap, const int* __restrict__ d_an,
                     float* __restrict__ out) {
    __shared__ float ssum[1024];
    __shared__ float scnt[1024];
    const int tid = threadIdx.x;

    float s = 0.0f, c = 0.0f;
    for (int i = tid; i < BATCH; i += 1024) {
        const int apb = d_ap[i];
        const int anb = d_an[i];
        const bool hasPos = (apb >= 0);
        const bool hasNeg = (anb != __float_as_int(1e30f));
        if (hasPos && hasNeg) {
            const float x = __int_as_float(apb) - __int_as_float(anb);
            const float sp = (x > 20.0f) ? x : log1pf(expf(x));
            s += sp;
            c += 1.0f;
        }
    }
    ssum[tid] = s;
    scnt[tid] = c;
    __syncthreads();
    #pragma unroll
    for (int off = 512; off > 0; off >>= 1) {
        if (tid < off) { ssum[tid] += ssum[tid + off]; scnt[tid] += scnt[tid + off]; }
        __syncthreads();
    }
    if (tid == 0) out[0] = ssum[0] / fmaxf(scnt[0], 1.0f);
}

// -------------------------------------------------------------------------
// Launcher
// -------------------------------------------------------------------------
extern "C" void kernel_launch(void* const* d_in, const int* in_sizes, int n_in,
                              void* d_out, int out_size, void* d_ws, size_t ws_size,
                              hipStream_t stream) {
    const float* feat  = (const float*)d_in[0];
    const int*   label = (const int*)d_in[1];

    char* ws = (char*)d_ws;
    _Float16* fh   = (_Float16*)ws;                                 // 4 MB
    float*    sq   = (float*)(ws + (size_t)BATCH * DIM * 2);
    int*      d_ap = (int*)(ws + (size_t)BATCH * DIM * 2 + BATCH * 4);
    int*      d_an = (int*)(ws + (size_t)BATCH * DIM * 2 + BATCH * 8);

    prep_kernel<<<BATCH, 128, 0, stream>>>(feat, fh, sq, d_ap, d_an);

    dim3 grid(BATCH / M_TILE, COL_SLICES);   // 64 x 8 workgroups
    miner_kernel<<<grid, 256, 0, stream>>>(fh, sq, label, d_ap, d_an);

    finalize_kernel<<<1, 1024, 0, stream>>>(d_ap, d_an, (float*)d_out);
}
